// TrackFormer_86114094285178
// MI455X (gfx1250) — matrix operand order repaired and checked by reading, not compiled, not run on hardware
//
#include <hip/hip_runtime.h>
#include <hip/hip_bf16.h>
#include <cstdint>
#include <cstddef>

// ---------------- problem constants (match reference) ----------------
constexpr int BN   = 8;          // batch
constexpr int BQ   = 900;        // queries
constexpr int BD   = 256;        // model dim
constexpr int HB_  = 200;
constexpr int WB_  = 200;
constexpr int NL   = 6;          // layers
constexpr int NHH  = 8;          // heads
constexpr int NP   = 4;          // sample points
constexpr int HD_  = 32;         // head dim
constexpr int FF_  = 512;
constexpr int NCLS_ = 10;
constexpr int ODIM_ = 18;
constexpr float THRESH_ = 0.5f;
constexpr float SCALE_  = 0.17677669529663687f;   // 1/sqrt(32)

constexpr int    NQ    = BN * BQ;                 // 7200 rows
constexpr size_t NQD_  = (size_t)BN * BQ * BD;    // 1,843,200

typedef __attribute__((ext_vector_type(16))) _Float16 v16h;
typedef __attribute__((ext_vector_type(8)))  float    v8f;

// =====================================================================
// f32-in / f32-out GEMM, f16 WMMA math, 32x32 register-blocked per wave.
// C = act(A @ W + bias)
//   A: [M x K] row-major, leading dim lda (runtime)
//   W: [K x Nout] row-major, leading dim LDW (compile-time -> immediate
//      offsets in the strided B loads, no 64-bit address chains)
//   C: [M x Nout], leading dim ldc
// act: 0 none, 1 relu, 2 sigmoid on cols < NCLS_ (det head)
// K must be a multiple of 32 (always 256 or 512 here).
// Out-of-range rows/cols: load pointers are clamped (B column j only
// affects D column j, so garbage never reaches stored outputs) and the
// stores are guarded; WMMA executes with full EXEC as the ISA requires.
// =====================================================================
template<int LDW>
__global__ __launch_bounds__(32)
void gemm_wmma(const float* __restrict__ A, int lda,
               const float* __restrict__ W,
               const float* __restrict__ bias,
               float* __restrict__ C, int ldc,
               int M, int K, int Nout, int act)
{
    const int lane = threadIdx.x;
    const int r16  = lane & 15;
    const int hi   = lane >> 4;          // 0 or 1
    const int mt   = blockIdx.x;
    const int nt   = blockIdx.y;

    const int row0 = mt * 32 + r16;
    const int row1 = row0 + 16;
    const int r0c  = (row0 < M) ? row0 : (M - 1);
    const int r1c  = (row1 < M) ? row1 : (M - 1);
    const float* ap0 = A + (size_t)r0c * lda;
    const float* ap1 = A + (size_t)r1c * lda;

    const int col0 = nt * 32 + r16;
    const int col1 = col0 + 16;
    const int c0c  = (col0 < Nout) ? col0 : (Nout - 1);
    const int c1c  = (col1 < Nout) ? col1 : (Nout - 1);
    const float* wp0 = W + (size_t)(hi * 16) * LDW + c0c;
    const float* wp1 = W + (size_t)(hi * 16) * LDW + c1c;

    v8f acc00 = {}, acc01 = {}, acc10 = {}, acc11 = {};

    for (int k0 = 0; k0 < K; k0 += 32) {
        v16h a0, a1, b0, b1;
        // A layout (16x32 f16): half i -> K = k0 + hi*16 + i
#pragma unroll
        for (int i = 0; i < 8; ++i) {
            a0[i]     = (_Float16)ap0[k0 + hi * 16 + i];
            a0[8 + i] = (_Float16)ap0[k0 + hi * 16 + 8 + i];
            a1[i]     = (_Float16)ap1[k0 + hi * 16 + i];
            a1[8 + i] = (_Float16)ap1[k0 + hi * 16 + 8 + i];
        }
        // B layout (32x16 f16): col = r16 lane slot, half i -> K = k0 + hi*16 + i
#pragma unroll
        for (int i = 0; i < 16; ++i) {
            b0[i] = (_Float16)wp0[i * LDW];
            b1[i] = (_Float16)wp1[i * LDW];
        }
        wp0 += 32 * LDW;
        wp1 += 32 * LDW;

        acc00 = __builtin_amdgcn_wmma_f32_16x16x32_f16(false, a0, false, b0, (short)0, acc00, false, false);
        acc01 = __builtin_amdgcn_wmma_f32_16x16x32_f16(false, a0, false, b1, (short)0, acc01, false, false);
        acc10 = __builtin_amdgcn_wmma_f32_16x16x32_f16(false, a1, false, b0, (short)0, acc10, false, false);
        acc11 = __builtin_amdgcn_wmma_f32_16x16x32_f16(false, a1, false, b1, (short)0, acc11, false, false);
    }

    // C layout: lane col = r16, component r -> row = rbase + hi*8 + r
    auto store_tile = [&](const v8f& acc, int rbase, int col) {
#pragma unroll
        for (int r = 0; r < 8; ++r) {
            int orow = rbase + hi * 8 + r;
            if (orow < M && col < Nout) {
                float v = acc[r] + (bias ? bias[col] : 0.0f);
                if (act == 1)      v = fmaxf(v, 0.0f);
                else if (act == 2 && col < NCLS_) v = 1.0f / (1.0f + __expf(-v));
                C[(size_t)orow * ldc + col] = v;
            }
        }
    };
    store_tile(acc00, mt * 32,      col0);
    store_tile(acc01, mt * 32,      col1);
    store_tile(acc10, mt * 32 + 16, col0);
    store_tile(acc11, mt * 32 + 16, col1);
}

// =====================================================================
// Fused flash attention (one wave per (n, h, 16-query tile)).
// Per 32-key chunk: two 16x16x32 score WMMAs; score tiles staged to LDS
// once; each lane then reads the 16 scores of ITS row (r16) at exactly
// its A-layout K slots, so row max/sum need only one __shfl_xor(.,16)
// with the partner lane, and exp(s-m) converts directly into the P
// operand (no second LDS round trip). Per-row rescale factors return to
// C-layout lanes via a 16-float LDS broadcast. Two WMMAs for P@V.
// =====================================================================
__global__ __launch_bounds__(32)
void attn_wmma(const float* __restrict__ qb, const float* __restrict__ kb,
               const float* __restrict__ vb, float* __restrict__ ob)
{
    __shared__ float sshm[16 * 32];
    __shared__ float scshm[16];
    const int lane = threadIdx.x;
    const int r16  = lane & 15;
    const int hi   = lane >> 4;
    const int qt = blockIdx.x, h = blockIdx.y, n = blockIdx.z;

    int qrow = qt * 16 + r16;
    int qcl  = (qrow < BQ) ? qrow : (BQ - 1);
    const float* qp = qb + ((size_t)(n * BQ + qcl)) * BD + h * HD_;

    v16h aQ;
#pragma unroll
    for (int i = 0; i < 8; ++i) {
        aQ[i]     = (_Float16)(qp[hi * 16 + i] * SCALE_);
        aQ[8 + i] = (_Float16)(qp[hi * 16 + 8 + i] * SCALE_);
    }

    v8f acc0 = {}, acc1 = {};
    float m = -1e30f, l = 0.0f;       // stats for row r16 (duplicated in both halves)

    const int nchunks = (BQ + 31) / 32;   // 29
    for (int c = 0; c < nchunks; ++c) {
        const int kbase = c * 32;
        const int key0 = kbase + r16;
        const int key1 = kbase + 16 + r16;
        const float* kp0 = kb + ((size_t)(n * BQ + (key0 < BQ ? key0 : BQ - 1))) * BD + h * HD_ + hi * 16;
        const float* kp1 = kb + ((size_t)(n * BQ + (key1 < BQ ? key1 : BQ - 1))) * BD + h * HD_ + hi * 16;
        v16h bK0, bK1;
#pragma unroll
        for (int i = 0; i < 16; ++i) {
            bK0[i] = (_Float16)kp0[i];
            bK1[i] = (_Float16)kp1[i];
        }
        v8f z = {};
        v8f s0 = __builtin_amdgcn_wmma_f32_16x16x32_f16(false, aQ, false, bK0, (short)0, z, false, false);
        v8f s1 = __builtin_amdgcn_wmma_f32_16x16x32_f16(false, aQ, false, bK1, (short)0, z, false, false);

        __syncthreads();
        // stage scores (C layout): row hi*8+r, col r16 (tile0) / 16+r16 (tile1)
#pragma unroll
        for (int r = 0; r < 8; ++r) {
            sshm[(hi * 8 + r) * 32 + r16]      = s0[r];
            sshm[(hi * 8 + r) * 32 + 16 + r16] = s1[r];
        }
        __syncthreads();

        // each lane: 16 scores of row r16 at its A-layout K slots
        // i<8: K = hi*8 + i ; i>=8: K = 16 + hi*8 + (i-8)
        float sv[16];
#pragma unroll
        for (int i = 0; i < 8; ++i) {
            sv[i]     = sshm[r16 * 32 + hi * 8 + i];
            sv[8 + i] = sshm[r16 * 32 + 16 + hi * 8 + i];
        }
        float mx = -1e30f;
#pragma unroll
        for (int i = 0; i < 16; ++i) {
            int K = (i < 8) ? (hi * 8 + i) : (16 + hi * 8 + (i - 8));
            if (kbase + K >= BQ) sv[i] = -1e30f;      // mask invalid keys
            mx = fmaxf(mx, sv[i]);
        }
        mx = fmaxf(mx, __shfl_xor(mx, 16, 32));       // combine with partner half
        float mnew = fmaxf(m, mx);
        float sc = __expf(m - mnew);
        float rs = 0.0f;
        v16h aP;
#pragma unroll
        for (int i = 0; i < 16; ++i) {
            float p = __expf(sv[i] - mnew);           // masked -> exp(-inf) = 0
            rs += p;
            aP[i] = (_Float16)p;
        }
        rs += __shfl_xor(rs, 16, 32);
        l = l * sc + rs;
        m = mnew;

        // broadcast per-row rescale to C-layout lanes
        if (hi == 0) scshm[r16] = sc;
        __syncthreads();
#pragma unroll
        for (int r = 0; r < 8; ++r) {
            float s = scshm[hi * 8 + r];
            acc0[r] *= s;
            acc1[r] *= s;
        }

        // V tiles in B layout: i<8 -> key kbase + hi*8 + i, i>=8 -> kbase + 16 + hi*8 + (i-8) ; col = out dim
        v16h bV0, bV1;
#pragma unroll
        for (int i = 0; i < 16; ++i) {
            int key = kbase + ((i < 8) ? (hi * 8 + i) : (16 + hi * 8 + (i - 8)));
            if (key >= BQ) key = BQ - 1;              // p==0 there anyway
            const float* vp = vb + ((size_t)(n * BQ + key)) * BD + h * HD_;
            bV0[i] = (_Float16)vp[r16];
            bV1[i] = (_Float16)vp[16 + r16];
        }
        acc0 = __builtin_amdgcn_wmma_f32_16x16x32_f16(false, aP, false, bV0, (short)0, acc0, false, false);
        acc1 = __builtin_amdgcn_wmma_f32_16x16x32_f16(false, aP, false, bV1, (short)0, acc1, false, false);
    }

    // distribute per-row 1/l to C-layout lanes and store
    __syncthreads();
    if (hi == 0) scshm[r16] = l;
    __syncthreads();
#pragma unroll
    for (int r = 0; r < 8; ++r) {
        int orow = qt * 16 + hi * 8 + r;
        if (orow < BQ) {
            float inv = 1.0f / scshm[hi * 8 + r];
            float* op = ob + ((size_t)(n * BQ + orow)) * BD + h * HD_;
            op[r16]      = acc0[r] * inv;
            op[16 + r16] = acc1[r] * inv;
        }
    }
}

// ---------------- residual + LayerNorm: y = LN(a + b) ----------------
__global__ __launch_bounds__(32)
void add_ln(const float* __restrict__ a, const float* __restrict__ b,
            const float* __restrict__ g, const float* __restrict__ be,
            float* __restrict__ y)
{
    const int row  = blockIdx.x;            // 0..NQ-1
    const int lane = threadIdx.x;
    const float* ap = a + (size_t)row * BD;
    const float* bp = b + (size_t)row * BD;
    float xv[8];
    float s = 0.0f;
#pragma unroll
    for (int j = 0; j < 8; ++j) { xv[j] = ap[j * 32 + lane] + bp[j * 32 + lane]; s += xv[j]; }
    for (int off = 1; off < 32; off <<= 1) s += __shfl_xor(s, off, 32);
    float mean = s * (1.0f / BD);
    float v = 0.0f;
#pragma unroll
    for (int j = 0; j < 8; ++j) { float d = xv[j] - mean; v += d * d; }
    for (int off = 1; off < 32; off <<= 1) v += __shfl_xor(v, off, 32);
    float rstd = rsqrtf(v * (1.0f / BD) + 1e-5f);
    float* yp = y + (size_t)row * BD;
#pragma unroll
    for (int j = 0; j < 8; ++j) {
        int d = j * 32 + lane;
        yp[d] = (xv[j] - mean) * rstd * g[d] + be[d];
    }
}

// ---------------- out = pos_emb broadcast ----------------
__global__ void init_out(const float* __restrict__ pe, float* __restrict__ o, int total)
{
    int i = blockIdx.x * blockDim.x + threadIdx.x;
    if (i < total) o[i] = pe[i % (BQ * BD)];
}

// ---------------- x = out + pos_emb (queries) ----------------
__global__ void add_query(const float* __restrict__ o, const float* __restrict__ pe,
                          float* __restrict__ x, int total)
{
    int i = blockIdx.x * blockDim.x + threadIdx.x;
    if (i < total) x[i] = o[i] + pe[i % (BQ * BD)];
}

// =====================================================================
// Deformable sampling: for each (n,q,h) compute the aw-softmax-weighted,
// bilinearly interpolated raw BEV row (256-dim) into agg. The per-head
// value projection (and its bias) is applied afterwards with WMMA GEMMs
// (interp and projection commute since both are linear; softmax and
// bilinear weights each sum to 1 so the bias passes through exactly).
// =====================================================================
__global__ __launch_bounds__(32)
void deform_gather(const float* __restrict__ bev, const float* __restrict__ offb,
                   const float* __restrict__ awb, float* __restrict__ agg)
{
    const int q = blockIdx.x, h = blockIdx.y, n = blockIdx.z;
    const int lane = threadIdx.x;
    const size_t nq = (size_t)n * BQ + q;

    // reference point (round(linspace(0, HB*WB-1, Q))[q])
    float fidx = rintf((float)q * ((float)(HB_ * WB_ - 1) / (float)(BQ - 1)));
    int ridx = (int)fidx;
    float rx = (float)(ridx % WB_);
    float ry = (float)(ridx / WB_);

    // softmax over P (uniform across lanes)
    const float* awp = awb + nq * (NHH * NP) + h * NP;
    float a0 = awp[0], a1 = awp[1], a2 = awp[2], a3 = awp[3];
    float mx = fmaxf(fmaxf(a0, a1), fmaxf(a2, a3));
    float e0 = __expf(a0 - mx), e1 = __expf(a1 - mx), e2 = __expf(a2 - mx), e3 = __expf(a3 - mx);
    float inv = 1.0f / (e0 + e1 + e2 + e3);
    float wgt[4] = { e0 * inv, e1 * inv, e2 * inv, e3 * inv };

    const float* offp = offb + nq * (NHH * NP * 2) + h * (NP * 2);
    float accv[8];
#pragma unroll
    for (int j = 0; j < 8; ++j) accv[j] = 0.0f;

    for (int p = 0; p < NP; ++p) {
        float lx = rx + offp[p * 2 + 0];          // OFFSET_SCALE == 1.0
        float ly = ry + offp[p * 2 + 1];
        lx = fminf(fmaxf(lx, 0.0f), (float)(WB_ - 1));
        ly = fminf(fmaxf(ly, 0.0f), (float)(HB_ - 1));
        float x0 = floorf(lx), y0 = floorf(ly);
        float x1 = fminf(x0 + 1.0f, (float)(WB_ - 1));
        float y1 = fminf(y0 + 1.0f, (float)(HB_ - 1));
        float wx = lx - x0, wy = ly - y0;
        int ix0 = (int)x0, ix1 = (int)x1, iy0 = (int)y0, iy1 = (int)y1;
        const size_t base = (size_t)n * (HB_ * WB_);
        const float* b00 = bev + (base + (size_t)iy0 * WB_ + ix0) * BD;
        const float* b10 = bev + (base + (size_t)iy0 * WB_ + ix1) * BD;
        const float* b01 = bev + (base + (size_t)iy1 * WB_ + ix0) * BD;
        const float* b11 = bev + (base + (size_t)iy1 * WB_ + ix1) * BD;
        float w00 = wgt[p] * (1.0f - wx) * (1.0f - wy);
        float w10 = wgt[p] * wx * (1.0f - wy);
        float w01 = wgt[p] * (1.0f - wx) * wy;
        float w11 = wgt[p] * wx * wy;
#pragma unroll
        for (int j = 0; j < 8; ++j) {
            int d = j * 32 + lane;
            accv[j] += w00 * b00[d] + w10 * b10[d] + w01 * b01[d] + w11 * b11[d];
        }
    }
    float* ap = agg + (nq * NHH + h) * (size_t)BD;
#pragma unroll
    for (int j = 0; j < 8; ++j) ap[j * 32 + lane] = accv[j];
}

// ---------------- finalize: copy output, detections, mask ----------------
__global__ void finalize_out(const float* __restrict__ outb, float* __restrict__ dout)
{
    const int DET_OFF  = (int)NQD_;
    const int MASK_OFF = DET_OFF + BN * BQ * ODIM_;
    const int LR_OFF   = MASK_OFF + BN * BQ;
    const int lastLR   = LR_OFF + (NL - 1) * BN * BQ * ODIM_;

    int i = blockIdx.x * blockDim.x + threadIdx.x;
    if (i < (int)NQD_) dout[i] = outb[i];
    if (i < BN * BQ * ODIM_) dout[DET_OFF + i] = dout[lastLR + i];
    if (i < BN * BQ) {
        float c0 = dout[lastLR + i * ODIM_];
        dout[MASK_OFF + i] = (c0 >= THRESH_) ? 1.0f : 0.0f;
    }
}

// =====================================================================
extern "C" void kernel_launch(void* const* d_in, const int* in_sizes, int n_in,
                              void* d_out, int out_size, void* d_ws, size_t ws_size,
                              hipStream_t stream)
{
    (void)in_sizes; (void)n_in; (void)out_size; (void)ws_size;

    const float* bev  = (const float*)d_in[0];
    const float* pe   = (const float*)d_in[1];
    const float* Wq   = (const float*)d_in[2];
    const float* bq   = (const float*)d_in[3];
    const float* Wk   = (const float*)d_in[4];
    const float* bk   = (const float*)d_in[5];
    const float* Wv   = (const float*)d_in[6];
    const float* bv   = (const float*)d_in[7];
    const float* Wo   = (const float*)d_in[8];
    const float* bo   = (const float*)d_in[9];
    const float* ln1g = (const float*)d_in[10];
    const float* ln1b = (const float*)d_in[11];
    const float* ln2g = (const float*)d_in[12];
    const float* ln2b = (const float*)d_in[13];
    const float* ln3g = (const float*)d_in[14];
    const float* ln3b = (const float*)d_in[15];
    const float* dvW  = (const float*)d_in[16];
    const float* dvb  = (const float*)d_in[17];
    const float* offW = (const float*)d_in[18];
    const float* offbW= (const float*)d_in[19];
    const float* awW  = (const float*)d_in[20];
    const float* awbW = (const float*)d_in[21];
    const float* doW  = (const float*)d_in[22];
    const float* dob  = (const float*)d_in[23];
    const float* f1W  = (const float*)d_in[24];
    const float* f1b  = (const float*)d_in[25];
    const float* f2W  = (const float*)d_in[26];
    const float* f2b  = (const float*)d_in[27];
    const float* dW1  = (const float*)d_in[28];
    const float* db1  = (const float*)d_in[29];
    const float* dW2  = (const float*)d_in[30];
    const float* db2  = (const float*)d_in[31];

    float* ws = (float*)d_ws;
    float* xb    = ws + 0 * NQD_;
    float* qb    = ws + 1 * NQD_;
    float* kb    = ws + 2 * NQD_;
    float* vb    = ws + 3 * NQD_;
    float* ob    = ws + 4 * NQD_;
    float* ab    = ws + 5 * NQD_;
    float* x2b   = ws + 6 * NQD_;
    float* dpre  = ws + 7 * NQD_;
    float* db    = ws + 8 * NQD_;
    float* x3b   = ws + 9 * NQD_;
    float* fb    = ws + 10 * NQD_;
    float* outb  = ws + 11 * NQD_;
    float* hidb  = ws + 12 * NQD_;
    float* hb    = ws + 13 * NQD_;                       // N*Q*FF = 2*NQD
    float* offbuf= ws + 15 * NQD_;                       // N*Q*64
    float* awbuf = ws + 15 * NQD_ + (size_t)BN * BQ * 64;
    float* aggb  = ws + 15 * NQD_ + (size_t)BN * BQ * 96; // N*Q*NH*D = 8*NQD

    float* dout  = (float*)d_out;
    const int LR_OFF = (int)NQD_ + BN * BQ * ODIM_ + BN * BQ;
    float* lr = dout + LR_OFF;

    auto gemm = [&](const float* A, int lda, const float* W, int ldw, const float* bias,
                    float* C, int ldc, int M, int K, int Nout, int act) {
        dim3 g((M + 31) / 32, (Nout + 31) / 32);
        switch (ldw) {
        case 256: gemm_wmma<256><<<g, dim3(32), 0, stream>>>(A, lda, W, bias, C, ldc, M, K, Nout, act); break;
        case 512: gemm_wmma<512><<<g, dim3(32), 0, stream>>>(A, lda, W, bias, C, ldc, M, K, Nout, act); break;
        case  64: gemm_wmma< 64><<<g, dim3(32), 0, stream>>>(A, lda, W, bias, C, ldc, M, K, Nout, act); break;
        case  32: gemm_wmma< 32><<<g, dim3(32), 0, stream>>>(A, lda, W, bias, C, ldc, M, K, Nout, act); break;
        default:  gemm_wmma< 18><<<g, dim3(32), 0, stream>>>(A, lda, W, bias, C, ldc, M, K, Nout, act); break;
        }
    };

    const int EW = 256;
    const int totNQD = (int)NQD_;

    init_out<<<(totNQD + EW - 1) / EW, EW, 0, stream>>>(pe, outb, totNQD);

    for (int i = 0; i < NL; ++i) {
        const float* Wq_i = Wq + (size_t)i * BD * BD;
        const float* Wk_i = Wk + (size_t)i * BD * BD;
        const float* Wv_i = Wv + (size_t)i * BD * BD;
        const float* Wo_i = Wo + (size_t)i * BD * BD;
        const float* dvW_i = dvW + (size_t)i * BD * BD;
        const float* doW_i = doW + (size_t)i * BD * BD;
        const float* f1W_i = f1W + (size_t)i * BD * FF_;
        const float* f2W_i = f2W + (size_t)i * FF_ * BD;
        const float* offW_i = offW + (size_t)i * BD * (NHH * NP * 2);
        const float* awW_i  = awW + (size_t)i * BD * (NHH * NP);

        // x = output + queries
        add_query<<<(totNQD + EW - 1) / EW, EW, 0, stream>>>(outb, pe, xb, totNQD);

        // ---- MHA ----
        gemm(xb, BD, Wq_i, BD, bq + i * BD, qb, BD, NQ, BD, BD, 0);
        gemm(xb, BD, Wk_i, BD, bk + i * BD, kb, BD, NQ, BD, BD, 0);
        gemm(xb, BD, Wv_i, BD, bv + i * BD, vb, BD, NQ, BD, BD, 0);
        attn_wmma<<<dim3((BQ + 15) / 16, NHH, BN), dim3(32), 0, stream>>>(qb, kb, vb, ob);
        gemm(ob, BD, Wo_i, BD, bo + i * BD, ab, BD, NQ, BD, BD, 0);
        add_ln<<<NQ, 32, 0, stream>>>(xb, ab, ln1g + i * BD, ln1b + i * BD, x2b);

        // ---- Deformable attention ----
        gemm(x2b, BD, offW_i, NHH * NP * 2, offbW + i * (NHH * NP * 2),
             offbuf, NHH * NP * 2, NQ, BD, NHH * NP * 2, 0);
        gemm(x2b, BD, awW_i, NHH * NP, awbW + i * (NHH * NP),
             awbuf, NHH * NP, NQ, BD, NHH * NP, 0);
        deform_gather<<<dim3(BQ, NHH, BN), dim3(32), 0, stream>>>(bev, offbuf, awbuf, aggb);
        for (int h = 0; h < NHH; ++h) {
            gemm(aggb + h * BD, NHH * BD,                 // A: agg[:, h, :]
                 dvW_i + h * HD_, BD,                     // W: dv_W[:, h*HD:(h+1)*HD]
                 dvb + (size_t)i * BD + h * HD_,          // bias slice
                 dpre + h * HD_, BD,                      // C: d_pre[:, h*HD:]
                 NQ, BD, HD_, 0);
        }
        gemm(dpre, BD, doW_i, BD, dob + i * BD, db, BD, NQ, BD, BD, 0);
        add_ln<<<NQ, 32, 0, stream>>>(x2b, db, ln2g + i * BD, ln2b + i * BD, x3b);

        // ---- FFN ----
        gemm(x3b, BD, f1W_i, FF_, f1b + i * FF_, hb, FF_, NQ, BD, FF_, 1);
        gemm(hb, FF_, f2W_i, BD, f2b + i * BD, fb, BD, NQ, FF_, BD, 0);
        add_ln<<<NQ, 32, 0, stream>>>(x3b, fb, ln3g + i * BD, ln3b + i * BD, outb);

        // ---- detection head (shared weights, per-layer output) ----
        gemm(outb, BD, dW1, BD, db1, hidb, BD, NQ, BD, BD, 1);
        gemm(hidb, BD, dW2, ODIM_, db2, lr + (size_t)i * BN * BQ * ODIM_, ODIM_,
             NQ, BD, ODIM_, 2);
    }

    finalize_out<<<(totNQD + EW - 1) / EW, EW, 0, stream>>>(outb, dout);
}
